// GroupedQueryAttention_40415642255494
// MI455X (gfx1250) — matrix operand order repaired and checked by reading, not compiled, not run on hardware
//
#include <hip/hip_runtime.h>

// MI455X / gfx1250 fused GQA:  Q/KV proj (bf16 WMMA) -> RoPE -> causal
// flash-attention (bf16 WMMA, fp32 online softmax) -> out proj (bf16 WMMA).
// Matmuls run on v_wmma_f32_16x16x32_bf16.  Global->LDS tile staging uses the
// CDNA5 async-copy path (global_load_async_to_lds_b128 / ASYNCcnt) with a
// double-buffered pipeline: stage i+1 DMA overlaps stage i WMMA, drained by a
// count-based s_wait_asynccnt (async loads complete in order).

typedef __attribute__((ext_vector_type(16))) __bf16 v16bf;
typedef __attribute__((ext_vector_type(8)))  float  v8f;

#define EMBED   2048
#define NHEADS  32
#define NKV     8
#define HD      64
#define SEQ     2048
#define BATCH   2
#define KVDIM   (NKV * HD)        // 512
#define KV2     (2 * KVDIM)       // 1024
#define TOKENS  (BATCH * SEQ)     // 4096

// ---------------------------------------------------------------- async copy
// GLOBAL_LOAD_ASYNC_TO_LDS_B128: per-lane global->LDS DMA, no VGPR round-trip.
// VDST = wave-relative LDS byte offset (low 32 bits of the generic pointer),
// VADDR = 64-bit global address, SADDR = off (GV mode).  Tracked by ASYNCcnt.
__device__ __forceinline__ void async_load_b128(const void* gptr, void* lptr) {
  unsigned lds = (unsigned)(size_t)lptr;
  unsigned long long ga = (unsigned long long)(size_t)gptr;
  asm volatile("global_load_async_to_lds_b128 %0, %1, off"
               :: "v"(lds), "v"(ga) : "memory");
}
// Wait until at most N async ops remain outstanding (in-order completion ->
// everything issued before the newest N has landed in LDS).
template <int N>
__device__ __forceinline__ void wait_async_le() {
  asm volatile("s_wait_asynccnt %0" :: "i"(N) : "memory");
}

// ---------------------------------------------------------------- fragments
// A-matrix (16x32, bf16): lane&15 = M row; element e -> K = (e&7) + 8*(lane>>4) + 16*(e>>3)
// Two contiguous 8-element runs per lane -> lowers to 2x ds_load_b128.
__device__ __forceinline__ v16bf frag_a_rm(const __bf16* p, int ld) {
  const int lane = threadIdx.x & 31;
  const int r = lane & 15, h = lane >> 4;
  v16bf f;
#pragma unroll
  for (int e = 0; e < 16; ++e) {
    const int k = (e & 7) + 8 * h + ((e >> 3) << 4);
    f[e] = p[r * ld + k];
  }
  return f;
}

// B-matrix (32x16) from an N-major / K-contiguous tile (row n holds K run):
// lane&15 = N col; element e -> K = (e&7) + 8*(lane>>4) + 16*(e>>3).  Two contiguous
// 8-elem runs per lane -> 2x ds_load_b128.
__device__ __forceinline__ v16bf frag_b_nk(const __bf16* p, int ld) {
  const int lane = threadIdx.x & 31;
  const int n = lane & 15, h = lane >> 4;
  v16bf f;
#pragma unroll
  for (int e = 0; e < 16; ++e) f[e] = p[n * ld + (e & 7) + 8 * h + ((e >> 3) << 4)];
  return f;
}

__device__ __forceinline__ v8f v8f_zero() {
  v8f z;
#pragma unroll
  for (int e = 0; e < 8; ++e) z[e] = 0.f;
  return z;
}

// ---------------------------------------------------------------- conversion
__global__ __launch_bounds__(256) void f32_to_bf16(const float* __restrict__ s,
                                                   __bf16* __restrict__ d, int n) {
  const int i = blockIdx.x * 256 + threadIdx.x;
  if (i < n) d[i] = (__bf16)s[i];
}

// ---------------------------------------------------------------- GEMM
// C[M,N] = A[M,K] * W[N,K]^T + bias ; A,W bf16 row-major (K contiguous), C fp32.
// Block tile 128x64, 8 waves each 32x32; K staged 64 wide, double-buffered.
__global__ __launch_bounds__(256) void gemm_bf16_nt(
    const __bf16* __restrict__ A, const __bf16* __restrict__ W,
    const float* __restrict__ bias, float* __restrict__ C,
    int M, int N, int K) {
  __shared__ __bf16 As[2][128][72];
  __shared__ __bf16 Ws[2][64][72];
  const int tid  = threadIdx.x;
  const int wave = tid >> 5, lane = tid & 31;
  const int wm = wave >> 1, wn = wave & 1;
  const int bm = blockIdx.y * 128, bn = blockIdx.x * 64;

  auto stage = [&](int ks, int bufi) {       // 6 async DMAs per wave
#pragma unroll
    for (int i = 0; i < 4; ++i) {            // A tile: 128x64 bf16
      const int u = tid + i * 256;
      const int row = u >> 3, col = (u & 7) * 8;
      async_load_b128(&A[(size_t)(bm + row) * K + ks * 64 + col], &As[bufi][row][col]);
    }
#pragma unroll
    for (int i = 0; i < 2; ++i) {            // W tile: 64x64 bf16
      const int u = tid + i * 256;
      const int row = u >> 3, col = (u & 7) * 8;
      async_load_b128(&W[(size_t)(bn + row) * K + ks * 64 + col], &Ws[bufi][row][col]);
    }
  };

  v8f acc[2][2];
#pragma unroll
  for (int i = 0; i < 2; ++i)
#pragma unroll
    for (int j = 0; j < 2; ++j) acc[i][j] = v8f_zero();

  const int nstage = K >> 6;
  stage(0, 0);
  for (int is = 0; is < nstage; ++is) {
    const int cur = is & 1;
    if (is + 1 < nstage) {
      stage(is + 1, cur ^ 1);                // DMA next stage behind compute
      if (is + 2 < nstage) {                 // warm L2 two stages ahead
        __builtin_prefetch(&A[(size_t)(bm + (tid >> 1)) * K + (is + 2) * 64 + (tid & 1) * 32], 0, 0);
        __builtin_prefetch(&W[(size_t)(bn + (tid & 63)) * K + (is + 2) * 64], 0, 0);
      }
      wait_async_le<6>();                    // stage `is` landed; `is+1` in flight
    } else {
      wait_async_le<0>();
    }
    __syncthreads();

#pragma unroll
    for (int kc = 0; kc < 2; ++kc) {
      v16bf af[2], bfr[2];
#pragma unroll
      for (int mi = 0; mi < 2; ++mi)
        af[mi] = frag_a_rm(&As[cur][wm * 32 + mi * 16][kc * 32], 72);
#pragma unroll
      for (int ni = 0; ni < 2; ++ni)
        bfr[ni] = frag_b_nk(&Ws[cur][wn * 32 + ni * 16][kc * 32], 72);
#pragma unroll
      for (int mi = 0; mi < 2; ++mi)
#pragma unroll
        for (int ni = 0; ni < 2; ++ni)
          acc[mi][ni] = __builtin_amdgcn_wmma_f32_16x16x32_bf16(
              false, af[mi], false, bfr[ni], (short)0, acc[mi][ni], false, false);
    }
    __syncthreads();                         // buffer reusable for stage is+2
  }

#pragma unroll
  for (int mi = 0; mi < 2; ++mi) {
    const int row0 = bm + wm * 32 + mi * 16 + 8 * (lane >> 4);
#pragma unroll
    for (int ni = 0; ni < 2; ++ni) {
      const int col = bn + wn * 32 + ni * 16 + (lane & 15);
      const float bv = bias[col];
      float* p = C + (size_t)row0 * N + col;
#pragma unroll
      for (int e = 0; e < 8; ++e)
        p[(size_t)e * N] = acc[mi][ni][e] + bv;
    }
  }
}

// ---------------------------------------------------------------- RoPE scatter
// q[B,T,2048], kv[B,T,1024] (fp32) -> Qb[B,H,T,64], Kb[B,Hk,T,64],
// Vt[B,Hk,64,T] (V stored pre-transposed so attention can async-DMA it).
__global__ __launch_bounds__(256) void rope_scatter(
    const float* __restrict__ qf, const float* __restrict__ kvf,
    __bf16* __restrict__ Qb, __bf16* __restrict__ Kb, __bf16* __restrict__ Vt) {
  const int bt = blockIdx.x;                 // token index
  const int b = bt / SEQ, t = bt % SEQ;
  const int tid = threadIdx.x;
  const float* qrow  = qf  + (size_t)bt * EMBED;
  const float* kvrow = kvf + (size_t)bt * KV2;

#pragma unroll
  for (int it = 0; it < 4; ++it) {           // Q: 32 heads * 32 rot pairs
    const int idx = tid + it * 256;
    const int hh = idx >> 5, i = idx & 31;
    const float freq = __powf(10000.f, -(float)(2 * i) * (1.f / 64.f));
    const float ang = (float)t * freq;
    const float c = __cosf(ang), s = __sinf(ang);
    const float x1 = qrow[hh * HD + i], x2 = qrow[hh * HD + i + 32];
    const size_t base = ((size_t)(b * NHEADS + hh) * SEQ + t) * HD;
    Qb[base + i]      = (__bf16)(x1 * c - x2 * s);
    Qb[base + i + 32] = (__bf16)(x2 * c + x1 * s);
  }
  {                                          // K: 8 heads * 32 rot pairs
    const int hh = tid >> 5, i = tid & 31;
    const float freq = __powf(10000.f, -(float)(2 * i) * (1.f / 64.f));
    const float ang = (float)t * freq;
    const float c = __cosf(ang), s = __sinf(ang);
    const float x1 = kvrow[hh * HD + i], x2 = kvrow[hh * HD + i + 32];
    const size_t base = ((size_t)(b * NKV + hh) * SEQ + t) * HD;
    Kb[base + i]      = (__bf16)(x1 * c - x2 * s);
    Kb[base + i + 32] = (__bf16)(x2 * c + x1 * s);
  }
#pragma unroll
  for (int it = 0; it < 2; ++it) {           // V transposed: [b, g, d, t]
    const int idx = tid + it * 256;
    const int hh = idx >> 6, d = idx & 63;
    Vt[((size_t)(b * NKV + hh) * HD + d) * SEQ + t] = (__bf16)kvrow[KVDIM + hh * HD + d];
  }
}

// ---------------------------------------------------------------- flash attention
// block = (b, head, 64-row q tile); 4 waves, each owns 16 q rows.
// K and (pre-transposed) V tiles double-buffer via async DMA; every WMMA
// operand read is a contiguous ds_load_b128.
__global__ __launch_bounds__(128) void attn_fwd(
    const __bf16* __restrict__ Qb, const __bf16* __restrict__ Kb,
    const __bf16* __restrict__ Vt, __bf16* __restrict__ ctx) {
  __shared__ __bf16 Qs[64][72];
  __shared__ __bf16 Ks[2][64][72];
  __shared__ __bf16 Vts[2][64][72];          // Vts[.][d][n] = V[n][d]
  __shared__ __bf16 Ps[64][72];

  const int qt = blockIdx.x, hq = blockIdx.y, b = blockIdx.z;
  const int g = hq >> 2;                     // KV head (rep = 4)
  const int tid = threadIdx.x;
  const int wave = tid >> 5, lane = tid & 31;
  const int wq = wave * 16;
  const int r = lane & 15, hh = lane >> 4;

  const __bf16* Qg  = Qb + ((size_t)(b * NHEADS + hq) * SEQ + qt * 64) * HD;
  const __bf16* Kg  = Kb + ((size_t)(b * NKV + g) * SEQ) * HD;
  const __bf16* Vtg = Vt + ((size_t)(b * NKV + g) * HD) * SEQ;

  auto stage = [&](int j, int bufi) {        // 8 async DMAs per wave
#pragma unroll
    for (int i = 0; i < 4; ++i) {
      const int u = tid + i * 128;
      const int row = u >> 3, col = (u & 7) * 8;
      async_load_b128(&Kg[((size_t)(j * 64 + row)) * HD + col], &Ks[bufi][row][col]);
      async_load_b128(&Vtg[(size_t)row * SEQ + j * 64 + col], &Vts[bufi][row][col]);
    }
  };

#pragma unroll
  for (int i = 0; i < 4; ++i) {              // Q tile 64x64, async DMA
    const int u = tid + i * 128;
    const int row = u >> 3, col = (u & 7) * 8;
    async_load_b128(&Qg[(size_t)row * HD + col], &Qs[row][col]);
  }
  stage(0, 0);                               // first K/V stage rides behind Q
  wait_async_le<8>();                        // Q landed; stage 0 still in flight
  __syncthreads();

  v16bf aQ[2];
  aQ[0] = frag_a_rm(&Qs[wq][0], 72);
  aQ[1] = frag_a_rm(&Qs[wq][32], 72);

  v8f acc[4];
#pragma unroll
  for (int dt = 0; dt < 4; ++dt) acc[dt] = v8f_zero();
  float row_max[8], row_sum[8];
#pragma unroll
  for (int e = 0; e < 8; ++e) { row_max[e] = -3.0e38f; row_sum[e] = 0.f; }

  const float scale = 0.125f;                // 1/sqrt(64)

  for (int j = 0; j <= qt; ++j) {
    const int cur = j & 1;
    if (j < qt) {
      stage(j + 1, cur ^ 1);                 // DMA next tile behind compute
      if (j + 2 <= qt)
        __builtin_prefetch(&Kg[((size_t)((j + 2) * 64 + (tid >> 1))) * HD + (tid & 1) * 32], 0, 0);
      wait_async_le<8>();                    // tile j landed; j+1 in flight
    } else {
      wait_async_le<0>();
    }
    __syncthreads();

    v8f s[4];                                // S = Q * K^T
#pragma unroll
    for (int nt = 0; nt < 4; ++nt) {
      v8f z = v8f_zero();
      z = __builtin_amdgcn_wmma_f32_16x16x32_bf16(
          false, aQ[0], false, frag_b_nk(&Ks[cur][nt * 16][0], 72), (short)0, z, false, false);
      z = __builtin_amdgcn_wmma_f32_16x16x32_bf16(
          false, aQ[1], false, frag_b_nk(&Ks[cur][nt * 16][32], 72), (short)0, z, false, false);
      s[nt] = z;
    }
    // scale + causal mask (diagonal tile only)
#pragma unroll
    for (int nt = 0; nt < 4; ++nt)
#pragma unroll
      for (int e = 0; e < 8; ++e) {
        float v = s[nt][e] * scale;
        if (j == qt && (nt * 16 + r) > (wq + e + 8 * hh)) v = -3.0e38f;
        s[nt][e] = v;
      }
    // online softmax (row = e + 8*hh across the 16-lane half)
    float mnew[8], sf[8], lsum[8];
#pragma unroll
    for (int e = 0; e < 8; ++e) {
      float m = fmaxf(fmaxf(s[0][e], s[1][e]), fmaxf(s[2][e], s[3][e]));
#pragma unroll
      for (int off = 1; off < 16; off <<= 1) m = fmaxf(m, __shfl_xor(m, off, 32));
      mnew[e] = fmaxf(row_max[e], m);
      sf[e] = __expf(row_max[e] - mnew[e]);
      lsum[e] = 0.f;
    }
#pragma unroll
    for (int nt = 0; nt < 4; ++nt)
#pragma unroll
      for (int e = 0; e < 8; ++e) {
        const float p = __expf(s[nt][e] - mnew[e]);
        s[nt][e] = p;
        lsum[e] += p;
      }
#pragma unroll
    for (int e = 0; e < 8; ++e) {
      float l = lsum[e];
#pragma unroll
      for (int off = 1; off < 16; off <<= 1) l += __shfl_xor(l, off, 32);
      row_sum[e] = row_sum[e] * sf[e] + l;
      row_max[e] = mnew[e];
    }
#pragma unroll
    for (int dt = 0; dt < 4; ++dt)
#pragma unroll
      for (int e = 0; e < 8; ++e) acc[dt][e] *= sf[e];

    // restage P (C-layout fp32 -> A-layout bf16) via per-wave LDS region;
    // LDS is in-order within a wave, so no barrier is needed here.
#pragma unroll
    for (int nt = 0; nt < 4; ++nt)
#pragma unroll
      for (int e = 0; e < 8; ++e)
        Ps[wq + e + 8 * hh][nt * 16 + r] = (__bf16)s[nt][e];

    v16bf aP[2];
    aP[0] = frag_a_rm(&Ps[wq][0], 72);
    aP[1] = frag_a_rm(&Ps[wq][32], 72);
#pragma unroll
    for (int dt = 0; dt < 4; ++dt) {
      acc[dt] = __builtin_amdgcn_wmma_f32_16x16x32_bf16(
          false, aP[0], false, frag_b_nk(&Vts[cur][dt * 16][0], 72), (short)0, acc[dt], false, false);
      acc[dt] = __builtin_amdgcn_wmma_f32_16x16x32_bf16(
          false, aP[1], false, frag_b_nk(&Vts[cur][dt * 16][32], 72), (short)0, acc[dt], false, false);
    }
    __syncthreads();                         // buffer reusable for stage j+2
  }

  __bf16* p = ctx + ((size_t)(b * SEQ + qt * 64 + wq + 8 * hh)) * EMBED + hq * HD + r;
#pragma unroll
  for (int dt = 0; dt < 4; ++dt)
#pragma unroll
    for (int e = 0; e < 8; ++e)
      p[(size_t)e * EMBED + dt * 16] = (__bf16)(acc[dt][e] / row_sum[e]);
}

// ---------------------------------------------------------------- launch
extern "C" void kernel_launch(void* const* d_in, const int* in_sizes, int n_in,
                              void* d_out, int out_size, void* d_ws, size_t ws_size,
                              hipStream_t stream) {
  const float* hs    = (const float*)d_in[0];
  const float* q_w   = (const float*)d_in[1];
  const float* q_b   = (const float*)d_in[2];
  const float* kv_w  = (const float*)d_in[3];
  const float* kv_b  = (const float*)d_in[4];
  const float* out_w = (const float*)d_in[5];
  const float* out_b = (const float*)d_in[6];

  char* ws = (char*)d_ws;
  size_t off = 0;
  auto alloc = [&](size_t bytes) -> void* {
    void* p = ws + off;
    off = (off + bytes + 255) & ~(size_t)255;
    return p;
  };
  __bf16* hidb = (__bf16*)alloc((size_t)TOKENS * EMBED * 2);
  __bf16* qwb  = (__bf16*)alloc((size_t)EMBED * EMBED * 2);
  __bf16* kvwb = (__bf16*)alloc((size_t)KV2 * EMBED * 2);
  __bf16* owb  = (__bf16*)alloc((size_t)EMBED * EMBED * 2);
  float*  qf   = (float*)alloc((size_t)TOKENS * EMBED * 4);
  float*  kvf  = (float*)alloc((size_t)TOKENS * KV2 * 4);
  __bf16* Qb   = (__bf16*)alloc((size_t)BATCH * NHEADS * SEQ * HD * 2);
  __bf16* Kb   = (__bf16*)alloc((size_t)BATCH * NKV * SEQ * HD * 2);
  __bf16* Vt   = (__bf16*)alloc((size_t)BATCH * NKV * SEQ * HD * 2);
  __bf16* ctxb = (__bf16*)alloc((size_t)TOKENS * EMBED * 2);

  auto cvt = [&](const float* s, __bf16* d, int n) {
    f32_to_bf16<<<(n + 255) / 256, 256, 0, stream>>>(s, d, n);
  };
  cvt(hs, hidb, TOKENS * EMBED);
  cvt(q_w, qwb, EMBED * EMBED);
  cvt(kv_w, kvwb, KV2 * EMBED);
  cvt(out_w, owb, EMBED * EMBED);

  gemm_bf16_nt<<<dim3(EMBED / 64, TOKENS / 128), 256, 0, stream>>>(
      hidb, qwb, q_b, qf, TOKENS, EMBED, EMBED);
  gemm_bf16_nt<<<dim3(KV2 / 64, TOKENS / 128), 256, 0, stream>>>(
      hidb, kvwb, kv_b, kvf, TOKENS, KV2, EMBED);

  rope_scatter<<<TOKENS, 256, 0, stream>>>(qf, kvf, Qb, Kb, Vt);

  attn_fwd<<<dim3(SEQ / 64, NHEADS, BATCH), 128, 0, stream>>>(Qb, Kb, Vt, ctxb);

  gemm_bf16_nt<<<dim3(EMBED / 64, TOKENS / 128), 256, 0, stream>>>(
      ctxb, owb, out_b, (float*)d_out, TOKENS, EMBED, EMBED);
}
